// attention_action_decoder_51908974739551
// MI455X (gfx1250) — matrix operand order repaired and checked by reading, not compiled, not run on hardware
//
#include <hip/hip_runtime.h>

// ---------------- CDNA5 WMMA types & helpers ----------------
typedef __attribute__((ext_vector_type(16))) __bf16 v16bf;
typedef __attribute__((ext_vector_type(8)))  float  v8f;

union FragU { v16bf v; unsigned int u[8]; };

// A-matrix fragment (16x32 bf16, wave32 layout):
// lane L: m = L%16, half = L/16; element i -> k = (i/8)*16 + (i%8) + half*8
// dword d holds elements (2d,2d+1): k = (d/4)*16 + (d%4)*2 + half*8 (+0/+1)
__device__ __forceinline__ v16bf load_a_frag(const __bf16* rowbase, int k0, int hf) {
  FragU f;
  const unsigned short* p = (const unsigned short*)rowbase + k0 + hf * 8;
#pragma unroll
  for (int d = 0; d < 8; ++d) {
    int k = ((d >> 2) << 4) + ((d & 3) << 1);
    f.u[d] = *(const unsigned int*)(p + k);
  }
  return f.v;
}

// B-matrix fragment (32x16 bf16): lane L: n = L%16, half = L/16;
// element i -> k = (i/8)*16 + (i%8) + half*8 (same as A).
__device__ __forceinline__ v16bf load_b_frag(const __bf16* rowbase, int k0, int hf) {
  FragU f;
  const unsigned short* p = (const unsigned short*)rowbase + k0 + hf * 8;
#pragma unroll
  for (int d = 0; d < 8; ++d) f.u[d] = *(const unsigned int*)(p + ((d >> 2) << 4) + ((d & 3) << 1));
  return f.v;
}

__device__ __forceinline__ v8f wmma_bf16(v16bf a, v16bf b, v8f c) {
  return __builtin_amdgcn_wmma_f32_16x16x32_bf16(false, a, false, b, (short)0, c, false, false);
}

// Opaque zero offset: loop-variant in the eyes of LICM (so loop-invariant weight
// loads are NOT hoisted out of the 64-step recurrence and spilled), but the
// pointer chain stays intact so the backend still infers GLOBAL address space
// (global_load_b128 with SADDR) instead of flat_load.
__device__ __forceinline__ int opaque_zero() {
  int ob = 0;
  asm volatile("" : "+v"(ob));
  return ob;
}

__device__ __forceinline__ float sigmoidf_(float x) { return 1.f / (1.f + __expf(-x)); }

// ---------------- problem constants ----------------
#define B_    256
#define T_    64
#define HID_  256
#define DINS  256   // 2*INS_H
#define LI_   128
#define LH_   256
#define EMB_  128
#define KQ    512   // HID + DINS
#define KBD   640   // dynamic K of W_bd: [z_c(256) | z_p(256) | x(128)]
#define BB    16    // batch rows per block
#define TPB   1024  // 32 waves

// ---------------- prep kernels ----------------
__global__ void cvt_bf16_k(const float* __restrict__ src, __bf16* __restrict__ dst, int n) {
  int i = blockIdx.x * 256 + threadIdx.x;
  if (i < n) dst[i] = (__bf16)src[i];
}

// Wbd_dyn[n][k]: k<512 -> W_bd cols 0..511 (z_c,z_p); k>=512 -> cols 1024..1151 (x)
__global__ void pack_wbd_k(const float* __restrict__ Wbd, __bf16* __restrict__ dst) {
  int i = blockIdx.x * 256 + threadIdx.x;
  if (i >= 256 * KBD) return;
  int nrow = i / KBD, k = i % KBD;
  int col = (k < 512) ? k : (512 + k);  // 512+k == 1024+(k-512)
  dst[i] = (__bf16)Wbd[(size_t)nrow * 1152 + col];
}

__global__ void pack_wout_k(const float* __restrict__ W, __bf16* __restrict__ dst) {
  int i = blockIdx.x * 256 + threadIdx.x;
  if (i >= 1008 * 256) return;
  int nrow = i / 256;
  dst[i] = (nrow < 1000) ? (__bf16)W[i] : (__bf16)0.f;
}

// static_bd[b][n] = b_bd[n] + sum_j (Wbd[n,512+j]+Wbd[n,640+j])*ini[b,j]
//                          + (Wbd[n,768+j]+Wbd[n,896+j])*cur[b,j]
// (attention over broadcast-identical rows is exactly the row itself)
__global__ void static_bd_k(const float* __restrict__ Wbd, const float* __restrict__ bbd,
                            const float* __restrict__ ini, const float* __restrict__ cur,
                            float* __restrict__ outsb) {
  int i = blockIdx.x * 256 + threadIdx.x;  // 65536 threads
  int b = i >> 8, n = i & 255;
  const float* wr = Wbd + (size_t)n * 1152;
  float a = bbd[n];
  for (int j = 0; j < 128; ++j) {
    a = fmaf(wr[512 + j] + wr[640 + j], ini[b * 128 + j], a);
    a = fmaf(wr[768 + j] + wr[896 + j], cur[b * 128 + j], a);
  }
  outsb[i] = a;
}

// ---------------- main recurrent kernel ----------------
// 16 blocks x 16 batch rows; 32 waves. Wave pairs: (w, w+16) split K (GEMM1/2/3)
// or gates i,f | g,o (GEMM4); waves (2b, 2b+1) pair up per batch for attention.
__global__ __launch_bounds__(TPB)
void decoder_kernel(const float* __restrict__ ins, const float* __restrict__ his,
                    const int* __restrict__ actions,
                    const float* __restrict__ h0, const float* __restrict__ c0,
                    const float* __restrict__ bih, const float* __restrict__ bhh,
                    const __bf16* __restrict__ Wc, const __bf16* __restrict__ Wp,
                    const __bf16* __restrict__ Wbd, const __bf16* __restrict__ Wih,
                    const __bf16* __restrict__ Whh, const __bf16* __restrict__ embb,
                    const float* __restrict__ static_bd,
                    __bf16* __restrict__ Hall) {
  __shared__ __bf16 qbuf[BB * KQ];    // [m][512]: cols 0..255 = h(bf16), 256..511 = z_c(bf16)
  __shared__ __bf16 hkb[BB * HID_];   // hk bf16
  __shared__ __bf16 zpb[BB * DINS];   // z_p bf16
  __shared__ float  vld[BB * HID_];   // attention query vector v
  __shared__ float  sld[BB * LH_];    // scores / split-K partials / gate staging (time-mux)

  const int tid = threadIdx.x;
  const int w = tid >> 5;        // wave 0..31
  const int lane = tid & 31;
  const int hf = lane >> 4;      // lane half
  const int l16 = lane & 15;
  const int b0 = blockIdx.x * BB;
  const int wg = w & 15;         // GEMM column-tile group
  const int kh = w >> 4;         // 0 = low half / gates i,f ; 1 = high half / gates g,o
  const int ab = w >> 1;         // attention batch (0..15)
  const int ap = w & 1;          // attention parity within pair

  // init h (bf16) into qbuf
  for (int idx = tid; idx < BB * HID_; idx += TPB) {
    int m = idx >> 8, j = idx & 255;
    qbuf[m * KQ + j] = (__bf16)h0[(size_t)(b0 + m) * HID_ + j];
  }
  // cell state lives in registers of waves 0..15: wave w owns columns jg = w*16+l16
  const int jg = wg * 16 + l16;
  float creg[8];
  float bA, bB;                  // kh0: bias(i),bias(f) ; kh1: bias(g),bias(o)
  if (kh == 0) {
#pragma unroll
    for (int r = 0; r < 8; ++r)
      creg[r] = c0[(size_t)(b0 + r + 8 * hf) * HID_ + jg];
    bA = bih[jg] + bhh[jg];
    bB = bih[256 + jg] + bhh[256 + jg];
  } else {
#pragma unroll
    for (int r = 0; r < 8; ++r) creg[r] = 0.f;
    bA = bih[512 + jg] + bhh[512 + jg];
    bB = bih[768 + jg] + bhh[768 + jg];
  }
  __syncthreads();

#pragma unroll 1
  for (int t = 0; t < T_; ++t) {
    // ---- GEMM1: v = h @ Wc^T (K=256, split K between wave pair) ----
    {
      v8f acc = {};
      const int n0 = wg * 16;
      const int ob = opaque_zero();
      const __bf16* wrow = &Wc[(size_t)(n0 + l16) * HID_ + ob];
      const int kb = kh * 128;
      v16bf a = load_a_frag(&qbuf[l16 * KQ], kb, hf);
      v16bf b = load_b_frag(wrow, kb, hf);
#pragma unroll
      for (int kk = 0; kk < 128; kk += 32) {
        v16bf ca = a, cb = b;
        if (kk + 32 < 128) {
          a = load_a_frag(&qbuf[l16 * KQ], kb + kk + 32, hf);
          b = load_b_frag(wrow, kb + kk + 32, hf);
        }
        acc = wmma_bf16(ca, cb, acc);
      }
      if (kh) {
#pragma unroll
        for (int r = 0; r < 8; ++r) sld[(r + 8 * hf) * LH_ + n0 + l16] = acc[r];
      }
      __syncthreads();
      if (!kh) {
#pragma unroll
        for (int r = 0; r < 8; ++r)
          vld[(r + 8 * hf) * HID_ + n0 + l16] = acc[r] + sld[(r + 8 * hf) * LH_ + n0 + l16];
      }
    }
    __syncthreads();

    // ---- attention over ins (wave pair (2b,2b+1) owns batch b0+ab) ----
    {
      const float* vrow = &vld[ab * HID_];
      const float* insb = ins + (size_t)(b0 + ab) * LI_ * DINS;
      // prefetch a his row for the later phase (L2 temporal hint)
      __builtin_prefetch(his + ((size_t)(b0 + ab) * LH_ + lane + 32 * ap) * DINS, 0, 1);
      float sc[2];
#pragma unroll
      for (int i = 0; i < 2; ++i) {
        const int l = lane + 32 * ap + 64 * i;
        const float* row = insb + (size_t)l * DINS;
        float a0 = 0, a1 = 0, a2 = 0, a3 = 0;
        for (int d2 = 0; d2 < DINS; d2 += 4) {
          a0 = fmaf(row[d2 + 0], vrow[d2 + 0], a0);
          a1 = fmaf(row[d2 + 1], vrow[d2 + 1], a1);
          a2 = fmaf(row[d2 + 2], vrow[d2 + 2], a2);
          a3 = fmaf(row[d2 + 3], vrow[d2 + 3], a3);
        }
        sc[i] = (a0 + a1) + (a2 + a3);
        sld[ab * LH_ + l] = sc[i];            // raw scores
      }
      __syncthreads();
      // both waves of the pair reduce deterministically over all 128 raw scores
      float vals[4], mx = -1e30f;
#pragma unroll
      for (int j = 0; j < 4; ++j) { vals[j] = sld[ab * LH_ + lane + 32 * j]; mx = fmaxf(mx, vals[j]); }
      for (int off = 16; off; off >>= 1) mx = fmaxf(mx, __shfl_xor(mx, off, 32));
      float sum = 0.f;
#pragma unroll
      for (int j = 0; j < 4; ++j) sum += __expf(vals[j] - mx);
      for (int off = 16; off; off >>= 1) sum += __shfl_xor(sum, off, 32);
      const float inv = 1.f / sum;
#pragma unroll
      for (int i = 0; i < 2; ++i) {
        const int l = lane + 32 * ap + 64 * i;
        sld[ab * LH_ + l] = __expf(sc[i] - mx) * inv;
      }
      __syncthreads();
      // context: 4 dims per lane per wave (pair covers all 256)
#pragma unroll
      for (int i = 0; i < 4; ++i) {
        const int d2 = lane + 32 * ap + 64 * i;
        float a0 = 0, a1 = 0;
        for (int l = 0; l < LI_; l += 2) {
          a0 = fmaf(sld[ab * LH_ + l + 0], insb[(size_t)(l + 0) * DINS + d2], a0);
          a1 = fmaf(sld[ab * LH_ + l + 1], insb[(size_t)(l + 1) * DINS + d2], a1);
        }
        qbuf[ab * KQ + HID_ + d2] = (__bf16)(a0 + a1);   // z_c
      }
    }
    __syncthreads();

    // ---- GEMM2: v = [h|z_c] @ Wp^T (K=512, split K) ----
    {
      v8f acc = {};
      const int n0 = wg * 16;
      const int ob = opaque_zero();
      const __bf16* wrow = &Wp[(size_t)(n0 + l16) * KQ + ob];
      const int kb = kh * 256;
      v16bf a = load_a_frag(&qbuf[l16 * KQ], kb, hf);
      v16bf b = load_b_frag(wrow, kb, hf);
#pragma unroll
      for (int kk = 0; kk < 256; kk += 32) {
        v16bf ca = a, cb = b;
        if (kk + 32 < 256) {
          a = load_a_frag(&qbuf[l16 * KQ], kb + kk + 32, hf);
          b = load_b_frag(wrow, kb + kk + 32, hf);
        }
        acc = wmma_bf16(ca, cb, acc);
      }
      if (kh) {
#pragma unroll
        for (int r = 0; r < 8; ++r) sld[(r + 8 * hf) * LH_ + n0 + l16] = acc[r];
      }
      __syncthreads();
      if (!kh) {
#pragma unroll
        for (int r = 0; r < 8; ++r)
          vld[(r + 8 * hf) * HID_ + n0 + l16] = acc[r] + sld[(r + 8 * hf) * LH_ + n0 + l16];
      }
    }
    __syncthreads();

    // ---- attention over his (256 rows, wave pair per batch) ----
    {
      const float* vrow = &vld[ab * HID_];
      const float* hisb = his + (size_t)(b0 + ab) * LH_ * DINS;
      float sc[4];
#pragma unroll
      for (int i = 0; i < 4; ++i) {
        const int l = lane + 32 * ap + 64 * i;
        const float* row = hisb + (size_t)l * DINS;
        float a0 = 0, a1 = 0, a2 = 0, a3 = 0;
        for (int d2 = 0; d2 < DINS; d2 += 4) {
          a0 = fmaf(row[d2 + 0], vrow[d2 + 0], a0);
          a1 = fmaf(row[d2 + 1], vrow[d2 + 1], a1);
          a2 = fmaf(row[d2 + 2], vrow[d2 + 2], a2);
          a3 = fmaf(row[d2 + 3], vrow[d2 + 3], a3);
        }
        sc[i] = (a0 + a1) + (a2 + a3);
        sld[ab * LH_ + l] = sc[i];
      }
      __syncthreads();
      float mx = -1e30f, vals[8];
#pragma unroll
      for (int j = 0; j < 8; ++j) { vals[j] = sld[ab * LH_ + lane + 32 * j]; mx = fmaxf(mx, vals[j]); }
      for (int off = 16; off; off >>= 1) mx = fmaxf(mx, __shfl_xor(mx, off, 32));
      float sum = 0.f;
#pragma unroll
      for (int j = 0; j < 8; ++j) sum += __expf(vals[j] - mx);
      for (int off = 16; off; off >>= 1) sum += __shfl_xor(sum, off, 32);
      const float inv = 1.f / sum;
#pragma unroll
      for (int i = 0; i < 4; ++i) {
        const int l = lane + 32 * ap + 64 * i;
        sld[ab * LH_ + l] = __expf(sc[i] - mx) * inv;
      }
      __syncthreads();
#pragma unroll
      for (int i = 0; i < 4; ++i) {
        const int d2 = lane + 32 * ap + 64 * i;
        float a0 = 0, a1 = 0;
        for (int l = 0; l < LH_; l += 2) {
          a0 = fmaf(sld[ab * LH_ + l + 0], hisb[(size_t)(l + 0) * DINS + d2], a0);
          a1 = fmaf(sld[ab * LH_ + l + 1], hisb[(size_t)(l + 1) * DINS + d2], a1);
        }
        zpb[ab * DINS + d2] = (__bf16)(a0 + a1);         // z_p
      }
    }
    __syncthreads();

    // ---- GEMM3: hk = tanh([z_c|z_p|x] @ Wbd_dyn^T + static_bd) (K=640, split K) ----
    {
      v8f acc = {};
      const int n0 = wg * 16;
      const int ob = opaque_zero();
      const __bf16* wrow = &Wbd[(size_t)(n0 + l16) * KBD + ob];
      const int aid = actions[(size_t)(b0 + l16) * T_ + t];  // embedding row for A row l16
      auto aload = [&](int kc) -> v16bf {
        if (kc < 256)      return load_a_frag(&qbuf[l16 * KQ + HID_], kc, hf);
        else if (kc < 512) return load_a_frag(&zpb[l16 * DINS], kc - 256, hf);
        else               return load_a_frag(&embb[(size_t)aid * EMB_ + ob], kc - 512, hf);
      };
      const int kb = kh * 320;
      v16bf a = aload(kb);
      v16bf b = load_b_frag(wrow, kb, hf);
#pragma unroll
      for (int kk = 0; kk < 320; kk += 32) {
        v16bf ca = a, cb = b;
        if (kk + 32 < 320) {
          a = aload(kb + kk + 32);
          b = load_b_frag(wrow, kb + kk + 32, hf);
        }
        acc = wmma_bf16(ca, cb, acc);
      }
      if (kh) {
#pragma unroll
        for (int r = 0; r < 8; ++r) sld[(r + 8 * hf) * LH_ + n0 + l16] = acc[r];
      }
      __syncthreads();
      if (!kh) {
#pragma unroll
        for (int r = 0; r < 8; ++r) {
          const int m = r + 8 * hf;
          float s = acc[r] + sld[m * LH_ + n0 + l16]
                  + static_bd[(size_t)(b0 + m) * HID_ + n0 + l16];
          hkb[m * HID_ + n0 + l16] = (__bf16)tanhf(s);
        }
      }
    }
    __syncthreads();

    // ---- GEMM4: gates = [hk|h] @ [Wih|Whh]^T, wave pair splits gates (i,f)|(g,o) ----
    {
      const int nb = wg * 16 + l16;
      const int gA = kh ? 512 : 0;    // g : i
      const int gB = kh ? 768 : 256;  // o : f
      const int ob = opaque_zero();
      const __bf16* wihA = &Wih[(size_t)(gA + nb) * HID_ + ob];
      const __bf16* wihB = &Wih[(size_t)(gB + nb) * HID_ + ob];
      const __bf16* whhA = &Whh[(size_t)(gA + nb) * HID_ + ob];
      const __bf16* whhB = &Whh[(size_t)(gB + nb) * HID_ + ob];
      auto aload = [&](int kc) -> v16bf {
        return (kc < 256) ? load_a_frag(&hkb[l16 * HID_], kc, hf)
                          : load_a_frag(&qbuf[l16 * KQ], kc - 256, hf);
      };
      auto bloadA = [&](int kc) -> v16bf {
        return (kc < 256) ? load_b_frag(wihA, kc, hf) : load_b_frag(whhA, kc - 256, hf);
      };
      auto bloadB = [&](int kc) -> v16bf {
        return (kc < 256) ? load_b_frag(wihB, kc, hf) : load_b_frag(whhB, kc - 256, hf);
      };
      v8f accA, accB;
#pragma unroll
      for (int r = 0; r < 8; ++r) { accA[r] = bA; accB[r] = bB; }
      v16bf a = aload(0), fA = bloadA(0), fB = bloadB(0);
#pragma unroll
      for (int kc = 0; kc < 512; kc += 32) {
        v16bf ca = a, cA = fA, cB = fB;
        if (kc + 32 < 512) {
          a = aload(kc + 32);
          fA = bloadA(kc + 32);
          fB = bloadB(kc + 32);
        }
        accA = wmma_bf16(ca, cA, accA);
        accB = wmma_bf16(ca, cB, accB);
      }
      // stage g-gate
      if (kh) {
#pragma unroll
        for (int r = 0; r < 8; ++r) sld[(r + 8 * hf) * LH_ + nb] = accA[r];
      }
      __syncthreads();
      float gv[8];
      if (!kh) {
#pragma unroll
        for (int r = 0; r < 8; ++r) gv[r] = sld[(r + 8 * hf) * LH_ + nb];
      }
      __syncthreads();
      // stage o-gate
      if (kh) {
#pragma unroll
        for (int r = 0; r < 8; ++r) sld[(r + 8 * hf) * LH_ + nb] = accB[r];
      }
      __syncthreads();
      if (!kh) {
#pragma unroll
        for (int r = 0; r < 8; ++r) {
          const int m = r + 8 * hf;
          const float iv = sigmoidf_(accA[r]);
          const float fv = sigmoidf_(accB[r]);
          const float gg = tanhf(gv[r]);
          const float oo = sigmoidf_(sld[m * LH_ + nb]);
          creg[r] = fv * creg[r] + iv * gg;
          const float hv = oo * tanhf(creg[r]);
          const __bf16 hb = (__bf16)hv;
          qbuf[m * KQ + nb] = hb;
          Hall[((size_t)(b0 + m) * T_ + t) * HID_ + nb] = hb;
        }
      }
    }
    __syncthreads();
  }
}

// ---------------- output projection: out[B*T,1000] = Hall @ Wout^T + b_out ----------------
#define NT_OUT 63  // 1008/16 N-tiles
__global__ __launch_bounds__(256)
void out_kernel(const __bf16* __restrict__ Hall, const __bf16* __restrict__ Wout,
                const float* __restrict__ bout, float* __restrict__ out) {
  const int lane = threadIdx.x & 31;
  const int wv = threadIdx.x >> 5;
  const int hf = lane >> 4;
  const int l16 = lane & 15;
  const int g = blockIdx.x * 8 + wv;         // global wave -> one 16x16 tile
  const int mt = g / NT_OUT, nt = g % NT_OUT;
  const int m0 = mt * 16, n0 = nt * 16;
  const int n = n0 + l16;
  const float bias = (n < 1000) ? bout[n] : 0.f;
  const __bf16* arow = &Hall[(size_t)(m0 + l16) * 256];
  const __bf16* brow = &Wout[(size_t)n * 256];
  v8f acc;
#pragma unroll
  for (int r = 0; r < 8; ++r) acc[r] = bias;
  v16bf a = load_a_frag(arow, 0, hf);
  v16bf b = load_b_frag(brow, 0, hf);
#pragma unroll
  for (int kc = 0; kc < 256; kc += 32) {
    v16bf ca = a, cb = b;
    if (kc + 32 < 256) {
      a = load_a_frag(arow, kc + 32, hf);
      b = load_b_frag(brow, kc + 32, hf);
    }
    acc = wmma_bf16(ca, cb, acc);
  }
#pragma unroll
  for (int r = 0; r < 8; ++r) {
    int m = m0 + r + 8 * hf;
    if (n < 1000) out[(size_t)m * 1000 + n] = acc[r];
  }
}

// ---------------- launch ----------------
extern "C" void kernel_launch(void* const* d_in, const int* in_sizes, int n_in,
                              void* d_out, int out_size, void* d_ws, size_t ws_size,
                              hipStream_t stream) {
  const float* ins  = (const float*)d_in[0];
  const float* his  = (const float*)d_in[1];
  const int*   act  = (const int*)d_in[2];
  const float* cur  = (const float*)d_in[3];
  const float* ini  = (const float*)d_in[4];
  const float* h0   = (const float*)d_in[6];
  const float* c0   = (const float*)d_in[7];
  const float* emb  = (const float*)d_in[8];
  const float* Wc   = (const float*)d_in[9];
  const float* Wp   = (const float*)d_in[10];
  const float* Wbd  = (const float*)d_in[15];
  const float* bbd  = (const float*)d_in[16];
  const float* Wih  = (const float*)d_in[17];
  const float* Whh  = (const float*)d_in[18];
  const float* bih  = (const float*)d_in[19];
  const float* bhh  = (const float*)d_in[20];
  const float* Wout = (const float*)d_in[21];
  const float* bout = (const float*)d_in[22];

  char* p = (char*)d_ws;
  auto take = [&](size_t bytes) { char* q = p; p += (bytes + 255) & ~(size_t)255; return q; };
  __bf16* wc_bf   = (__bf16*)take((size_t)256 * 256 * 2);
  __bf16* wp_bf   = (__bf16*)take((size_t)256 * 512 * 2);
  __bf16* wbd_bf  = (__bf16*)take((size_t)256 * KBD * 2);
  __bf16* wih_bf  = (__bf16*)take((size_t)1024 * 256 * 2);
  __bf16* whh_bf  = (__bf16*)take((size_t)1024 * 256 * 2);
  __bf16* wout_bf = (__bf16*)take((size_t)1008 * 256 * 2);
  __bf16* emb_bf  = (__bf16*)take((size_t)1000 * 128 * 2);
  float*  sbd     = (float*)take((size_t)256 * 256 * 4);
  __bf16* Hall    = (__bf16*)take((size_t)256 * 64 * 256 * 2);

  cvt_bf16_k<<<(65536 + 255) / 256, 256, 0, stream>>>(Wc, wc_bf, 65536);
  cvt_bf16_k<<<(131072 + 255) / 256, 256, 0, stream>>>(Wp, wp_bf, 131072);
  pack_wbd_k<<<(256 * KBD + 255) / 256, 256, 0, stream>>>(Wbd, wbd_bf);
  cvt_bf16_k<<<(262144 + 255) / 256, 256, 0, stream>>>(Wih, wih_bf, 262144);
  cvt_bf16_k<<<(262144 + 255) / 256, 256, 0, stream>>>(Whh, whh_bf, 262144);
  pack_wout_k<<<(1008 * 256 + 255) / 256, 256, 0, stream>>>(Wout, wout_bf);
  cvt_bf16_k<<<(128000 + 255) / 256, 256, 0, stream>>>(emb, emb_bf, 128000);
  static_bd_k<<<256, 256, 0, stream>>>(Wbd, bbd, ini, cur, sbd);

  decoder_kernel<<<B_ / BB, TPB, 0, stream>>>(ins, his, act, h0, c0, bih, bhh,
                                              wc_bf, wp_bf, wbd_bf, wih_bf, whh_bf,
                                              emb_bf, sbd, Hall);

  // 1024 M-tiles * 63 N-tiles = 64512 waves / 8 waves per block = 8064 blocks
  out_kernel<<<8064, 256, 0, stream>>>(Hall, wout_bf, bout, (float*)d_out);
}
